// SelfAttention_55259049230981
// MI455X (gfx1250) — matrix operand order repaired and checked by reading, not compiled, not run on hardware
//
#include <hip/hip_runtime.h>
#include <hip/hip_bf16.h>

// ---------------------------------------------------------------------------
// SAGAN self-attention on MI455X (gfx1250, wave32, bf16 WMMA, TDM staging).
// Flash-attention reformulation:  Q = g^T (N x 64), K = f^T (N x 64),
// V = hv^T (N x 512);  O = softmax_k(Q K^T) V;  out[c][j] = g0*O[j][c]+x[c][j].
// Kernel 2 splits the 512 value channels over 2 blocks per query tile and
// double-buffers the V tile through the Tensor Data Mover.
// ---------------------------------------------------------------------------

typedef __attribute__((ext_vector_type(16))) __bf16 v16bf;
typedef __attribute__((ext_vector_type(8)))  float  v8f;
typedef __attribute__((ext_vector_type(4)))  unsigned int v4u;
typedef __attribute__((ext_vector_type(8)))  int v8i;
typedef __attribute__((ext_vector_type(4)))  int v4i;

union AFrag { v16bf v; uint4 q[2]; __bf16 e[16]; };
union Acc   { v8f  v; float4 q[2]; float e[8]; };

#define N_TOK 4096
#define C_IN  512
#define CQ    64
#define BATCH 4
#define NT_TILES (N_TOK / 32)
#define CSLICE 256                 // channels per attention block
#define VBUF (CSLICE * 40)         // one V buffer: 256 chans x stride 40

#if defined(__gfx1250__) && __has_builtin(__builtin_amdgcn_tensor_load_to_lds)
#define HAVE_TDM 1
#else
#define HAVE_TDM 0
#endif

// ---------------------------------------------------------------------------
// Kernel 1: projections.
//   FG[b][token][0:128]  = [ f(token) | g(token) ]   (bf16, token-major)
//   Vt[b][c][token]      = hv[c][token]              (bf16, channel-major)
// ---------------------------------------------------------------------------
__global__ __launch_bounds__(256) void proj_qkv_kernel(
    const float* __restrict__ x,  const float* __restrict__ Wq,
    const float* __restrict__ Wk, const float* __restrict__ Wv,
    __bf16* __restrict__ FG, __bf16* __restrict__ Vt)
{
  __shared__ __align__(16) __bf16 a_lds[128 * 40];   // [token][c]
  __shared__ __align__(16) __bf16 bT_lds[128 * 40];  // [o][c]

  const int t    = threadIdx.x;
  const int lane = t & 31;
  const int wave = t >> 5;
  const int hh   = (lane >> 4);
  const int j0   = blockIdx.x * 128;
  const int q    = blockIdx.y;         // 0: FG cols, 1..4: V channel blocks
  const int b    = blockIdx.z;
  const float* xb = x + (size_t)b * C_IN * N_TOK;

  Acc acc[2][4];
  for (int mi = 0; mi < 2; ++mi)
    for (int ni = 0; ni < 4; ++ni) acc[mi][ni].v = {};

  const int m_base = (wave & 3) * 32;
  const int n_base = (wave >> 2) * 64;

  for (int k0 = 0; k0 < C_IN; k0 += 32) {
    for (int i = 0; i < 8; ++i) {
      int idx = i * 256 + t;
      int token = idx & 127, c = (idx >> 7) * 2;
      float f0 = xb[(size_t)(k0 + c) * N_TOK + j0 + token];
      float f1 = xb[(size_t)(k0 + c + 1) * N_TOK + j0 + token];
      union { __bf16 h[2]; unsigned u; } pk;
      pk.h[0] = (__bf16)f0; pk.h[1] = (__bf16)f1;
      *(unsigned*)&a_lds[token * 40 + c] = pk.u;
    }
    for (int i = 0; i < 4; ++i) {
      int idx = i * 256 + t;
      int o = idx >> 3, c4 = (idx & 7) * 4;
      int og = q * 128 + o;
      const float* wrow = (og < CQ)     ? (Wq + (size_t)og * C_IN)
                        : (og < 2 * CQ) ? (Wk + (size_t)(og - CQ) * C_IN)
                                        : (Wv + (size_t)(og - 2 * CQ) * C_IN);
      float4 wv = *(const float4*)(wrow + k0 + c4);
      union { __bf16 h[4]; unsigned long long u; } pk;
      pk.h[0] = (__bf16)wv.x; pk.h[1] = (__bf16)wv.y;
      pk.h[2] = (__bf16)wv.z; pk.h[3] = (__bf16)wv.w;
      *(unsigned long long*)&bT_lds[o * 40 + c4] = pk.u;
    }
    __syncthreads();

    AFrag af[2], bfr[4];
    for (int mi = 0; mi < 2; ++mi) {
      const __bf16* pa = a_lds + (m_base + mi * 16 + (lane & 15)) * 40 + (hh << 3);
      af[mi].q[0] = *(const uint4*)pa;
      af[mi].q[1] = *(const uint4*)(pa + 16);
    }
    for (int ni = 0; ni < 4; ++ni) {
      const __bf16* pb = bT_lds + (n_base + ni * 16 + (lane & 15)) * 40 + (hh << 3);
      bfr[ni].q[0] = *(const uint4*)pb;
      bfr[ni].q[1] = *(const uint4*)(pb + 16);
    }
    for (int mi = 0; mi < 2; ++mi)
      for (int ni = 0; ni < 4; ++ni)
        acc[mi][ni].v = __builtin_amdgcn_wmma_f32_16x16x32_bf16(
            false, af[mi].v, false, bfr[ni].v, (short)0, acc[mi][ni].v,
            false, false);
    __syncthreads();
  }

  if (q == 0) {
    __bf16* FGb = FG + (size_t)b * N_TOK * 128;
    for (int mi = 0; mi < 2; ++mi)
      for (int ni = 0; ni < 4; ++ni)
        for (int v = 0; v < 8; ++v) {
          int token = j0 + m_base + mi * 16 + v + (hh << 3);
          int col = n_base + ni * 16 + (lane & 15);
          FGb[(size_t)token * 128 + col] = (__bf16)acc[mi][ni].e[v];
        }
  } else {
    __bf16* Vtb = Vt + (size_t)b * C_IN * N_TOK;
    for (int mi = 0; mi < 2; ++mi)
      for (int ni = 0; ni < 4; ++ni) {
        int c = (q - 1) * 128 + n_base + ni * 16 + (lane & 15);
        int tb = j0 + m_base + mi * 16 + (hh << 3);
        union { __bf16 h[8]; uint4 u; } pk;
        for (int v = 0; v < 8; ++v) pk.h[v] = (__bf16)acc[mi][ni].e[v];
        *(uint4*)(Vtb + (size_t)c * N_TOK + tb) = pk.u;
      }
  }
}

// ---------------------------------------------------------------------------
// TDM issue helper: load one 256-chan x 32-key tile of Vt into LDS with
// padded rows (pad 4 DWORDs per 16 DWORDs -> stride 40 bf16 elements).
// ---------------------------------------------------------------------------
#if HAVE_TDM
__device__ __forceinline__ void tdm_load_vtile(const __bf16* gsrc,
                                               unsigned lds_off) {
  unsigned long long ga = (unsigned long long)gsrc;
  v4u g0 = { 1u,                                      // count=1
             lds_off,
             (unsigned)(ga & 0xffffffffu),
             (unsigned)(((ga >> 32) & 0x1ffffffu) | (2u << 30)) };
  v8i g1 = { (int)((1u << 16)            // data_size = 2 bytes
                 | (1u << 20)            // pad_enable
                 | (3u << 22)            // pad_interval: 16 DWORDs
                 | (3u << 25)),          // pad_amount:   4 DWORDs
             (int)(4096u << 16),         // tensor_dim0 = 4096
             (int)(512u << 16),          // tensor_dim1 = 512
             (int)(32u << 16),           // tile_dim0 = 32 keys
             (int)CSLICE,                // tile_dim1 = 256 chans, tile_dim2=0
             (int)4096,                  // tensor_dim0_stride
             0, 0 };
  v4i g2 = { 0, 0, 0, 0 };
  v4i g3 = { 0, 0, 0, 0 };
#if __clang_major__ >= 23
  v8i g4 = { 0, 0, 0, 0, 0, 0, 0, 0 };
  __builtin_amdgcn_tensor_load_to_lds(g0, g1, g2, g3, g4, 0);
#else
  __builtin_amdgcn_tensor_load_to_lds(g0, g1, g2, g3, 0);
#endif
}
#endif

// ---------------------------------------------------------------------------
// Kernel 2: flash attention. Grid (N/64, 2 channel slices, B).
// 64 queries / block, 32-key tiles, 8 waves; wave owns 32 value channels.
// V tiles double-buffered through the Tensor Data Mover.
// ---------------------------------------------------------------------------
__global__ __launch_bounds__(256) void attn_kernel(
    const __bf16* __restrict__ FG, const __bf16* __restrict__ Vt,
    const float* __restrict__ x, const float* __restrict__ gamma,
    float* __restrict__ out)
{
  __shared__ __align__(16) __bf16 vT_lds[2 * VBUF];  // 2 x [256 chan][40] 40960 B
  __shared__ __align__(16) __bf16 q_lds[64 * 72];    //  9216 B
  __shared__ __align__(16) __bf16 k_lds[32 * 72];    //  4608 B
  __shared__ __align__(16) float  sp_lds[64 * 32];   //  8192 B (S f32 / P bf16)
  __shared__ __align__(16) float  row_max[64];
  __shared__ __align__(16) float  row_sum[64];
  __shared__ __align__(16) float  row_corr[64];

  __bf16* p_ov = (__bf16*)sp_lds;

  const int t    = threadIdx.x;
  const int lane = t & 31;
  const int wave = t >> 5;
  const int hh   = (lane >> 4);
  const int j0   = blockIdx.x * 64;
  const int cs   = blockIdx.y;         // channel slice: 0 or 1
  const int b    = blockIdx.z;
  const __bf16* FGb = FG + (size_t)b * N_TOK * 128;
  const __bf16* Vts = Vt + ((size_t)b * C_IN + (size_t)cs * CSLICE) * N_TOK;

  // Q tile = g projection (cols 64..127)
  for (int i = 0; i < 2; ++i) {
    int idx = i * 256 + t;
    int r = idx >> 3, c8 = (idx & 7) * 8;
    *(uint4*)&q_lds[r * 72 + c8] =
        *(const uint4*)(FGb + (size_t)(j0 + r) * 128 + CQ + c8);
  }
  if (t < 64) { row_max[t] = -3.0e38f; row_sum[t] = 0.0f; }

  Acc O[4][2];
  for (int a = 0; a < 4; ++a)
    for (int c = 0; c < 2; ++c) O[a][c].v = {};

  const int cb  = wave * 32;           // channel base within slice
  const int smb = (wave & 3) * 16;
  const int snb = (wave >> 2) * 16;

#if HAVE_TDM
  if (wave == 0)                       // preload tile 0 into buffer 0
    tdm_load_vtile(Vts, (unsigned)(unsigned long long)&vT_lds[0]);
#endif
  __syncthreads();

  for (int kt = 0; kt < NT_TILES; ++kt) {
    const int n0 = kt * 32;
    const __bf16* vbuf = vT_lds + (kt & 1) * VBUF;

#if HAVE_TDM
    if (wave == 0) {
      if (kt + 1 < NT_TILES) {         // DMA next tile into the other buffer
        tdm_load_vtile(Vts + n0 + 32,
                       (unsigned)(unsigned long long)&vT_lds[((kt + 1) & 1) * VBUF]);
        __builtin_amdgcn_s_wait_tensorcnt(1);   // current tile has landed
      } else {
        __builtin_amdgcn_s_wait_tensorcnt(0);
      }
    }
#else
    for (int i = 0; i < 4; ++i) {      // synchronous fallback
      int idx = i * 256 + t;
      int c = idx >> 2, k8 = (idx & 3) * 8;
      *(uint4*)&vT_lds[(kt & 1) * VBUF + c * 40 + k8] =
          *(const uint4*)(Vts + (size_t)c * N_TOK + n0 + k8);
    }
#endif
    // K tile = f projection (cols 0..63)
    {
      int r = t >> 3, c8 = (t & 7) * 8;
      *(uint4*)&k_lds[r * 72 + c8] =
          *(const uint4*)(FGb + (size_t)(n0 + r) * 128 + c8);
      if (kt + 1 < NT_TILES)
        __builtin_prefetch(FGb + (size_t)(n0 + 32 + r) * 128 + c8, 0, 1);
    }
    __syncthreads();

    // S = Q K^T : one 16x16 fragment per wave
    Acc sacc; sacc.v = {};
    for (int ks = 0; ks < 64; ks += 32) {
      AFrag qa, kb;
      const __bf16* pq = q_lds + (smb + (lane & 15)) * 72 + ks + (hh << 3);
      qa.q[0] = *(const uint4*)pq;
      qa.q[1] = *(const uint4*)(pq + 16);
      const __bf16* pk = k_lds + (snb + (lane & 15)) * 72 + ks + (hh << 3);
      kb.q[0] = *(const uint4*)pk;
      kb.q[1] = *(const uint4*)(pk + 16);
      sacc.v = __builtin_amdgcn_wmma_f32_16x16x32_bf16(
          false, qa.v, false, kb.v, (short)0, sacc.v, false, false);
    }
    for (int v = 0; v < 8; ++v)
      sp_lds[(smb + v + (hh << 3)) * 32 + snb + (lane & 15)] = sacc.e[v];
    __syncthreads();

    // Online softmax over keys, phase-split so bf16 P reuses the f32 S buffer
    union { uint4 q[4]; __bf16 h[32]; } pk;
    if (t < 64) {
      union { float4 q[8]; float f[32]; } srow;
      const float4* sp4 = (const float4*)(sp_lds + t * 32);
      for (int i = 0; i < 8; ++i) srow.q[i] = sp4[i];
      float mx = row_max[t];
      for (int n = 0; n < 32; ++n) mx = fmaxf(mx, srow.f[n]);
      float corr = __expf(row_max[t] - mx);
      float s = row_sum[t] * corr;
      for (int n = 0; n < 32; ++n) {
        float e = __expf(srow.f[n] - mx);
        s += e;
        pk.h[n] = (__bf16)e;
      }
      row_max[t] = mx; row_sum[t] = s; row_corr[t] = corr;
    }
    __syncthreads();
    if (t < 64) {
      uint4* pd = (uint4*)(p_ov + t * 32);
      for (int i = 0; i < 4; ++i) pd[i] = pk.q[i];
    }
    __syncthreads();

    // Rescale O, then O += P * V (one K=32 WMMA step per fragment)
    for (int mf = 0; mf < 4; ++mf) {
      Acc sc;
      const float* rc = row_corr + mf * 16 + (hh << 3);
      sc.q[0] = *(const float4*)rc;
      sc.q[1] = *(const float4*)(rc + 4);

      AFrag pa;
      const __bf16* pp = p_ov + (mf * 16 + (lane & 15)) * 32 + (hh << 3);
      pa.q[0] = *(const uint4*)pp;
      pa.q[1] = *(const uint4*)(pp + 16);

      for (int nf = 0; nf < 2; ++nf) {
        O[mf][nf].v *= sc.v;
        AFrag vb;
        const __bf16* pv = vbuf + (cb + nf * 16 + (lane & 15)) * 40 + (hh << 3);
        vb.q[0] = *(const uint4*)pv;
        vb.q[1] = *(const uint4*)(pv + 16);
        O[mf][nf].v = __builtin_amdgcn_wmma_f32_16x16x32_bf16(
            false, pa.v, false, vb.v, (short)0, O[mf][nf].v, false, false);
      }
    }
    __syncthreads();
  }

  // Epilogue: out[b][c][j] = g0 * O[j][c] / row_sum[j] + x[b][c][j]
  const float g0 = gamma[0];
  const float* xb = x + (size_t)b * C_IN * N_TOK;
  float* ob = out + (size_t)b * C_IN * N_TOK;
  for (int mf = 0; mf < 4; ++mf) {
    Acc scale;
    const float* rs = row_sum + mf * 16 + (hh << 3);
    scale.q[0] = *(const float4*)rs;
    scale.q[1] = *(const float4*)(rs + 4);
    for (int v = 0; v < 8; ++v) scale.e[v] = g0 / scale.e[v];
    int jb = j0 + mf * 16 + (hh << 3);
    for (int nf = 0; nf < 2; ++nf) {
      int c = cs * CSLICE + cb + nf * 16 + (lane & 15);
      const float* xp = xb + (size_t)c * N_TOK + jb;
      Acc res;
      res.q[0] = *(const float4*)xp;
      res.q[1] = *(const float4*)(xp + 4);
      res.v = O[mf][nf].v * scale.v + res.v;
      float* op = ob + (size_t)c * N_TOK + jb;
      *(float4*)op = res.q[0];
      *(float4*)(op + 4) = res.q[1];
    }
  }
}

// ---------------------------------------------------------------------------
extern "C" void kernel_launch(void* const* d_in, const int* in_sizes, int n_in,
                              void* d_out, int out_size, void* d_ws, size_t ws_size,
                              hipStream_t stream) {
  (void)in_sizes; (void)n_in; (void)out_size; (void)ws_size;
  const float* x     = (const float*)d_in[0];
  const float* Wq    = (const float*)d_in[1];
  const float* Wk    = (const float*)d_in[2];
  const float* Wv    = (const float*)d_in[3];
  const float* gamma = (const float*)d_in[4];
  float* out = (float*)d_out;

  __bf16* FG = (__bf16*)d_ws;                          //  4 MB
  __bf16* Vt = FG + (size_t)BATCH * N_TOK * 128;       // 16 MB

  dim3 g1(N_TOK / 128, 5, BATCH);
  proj_qkv_kernel<<<g1, 256, 0, stream>>>(x, Wq, Wk, Wv, FG, Vt);

  dim3 g2(N_TOK / 64, 2, BATCH);
  attn_kernel<<<g2, 256, 0, stream>>>(FG, Vt, x, gamma, out);
}
